// GCN_in_G_30803505447133
// MI455X (gfx1250) — matrix operand order repaired and checked by reading, not compiled, not run on hardware
//
#include <hip/hip_runtime.h>
#include <hip/hip_bf16.h>

typedef __attribute__((ext_vector_type(16))) __bf16 v16bf;
typedef __attribute__((ext_vector_type(8)))  float  v8f;

#define LD_X   192      // padded feature stride for node buffers (>= 160, mult of 32)
#define NNODES 50000
#define NEDGES 800000
#define NWP    256
#define CORE   160
#define HID    2048
#define KCH    256      // K-chunk staged in LDS for the pair kernel

// ---------------------------------------------------------------------------
// helpers
// ---------------------------------------------------------------------------
__device__ __forceinline__ float sigmoidf_(float x) {
  return 1.0f / (1.0f + __expf(-x));
}

__device__ __forceinline__ v8f wmma_bf16(v16bf a, v16bf b, v8f c) {
  // D = A(16x32 bf16) x B(32x16 bf16) + C(16x16 f32)
  return __builtin_amdgcn_wmma_f32_16x16x32_bf16(
      /*neg_a=*/false, a, /*neg_b=*/false, b,
      /*c_mod=*/(short)0, c, /*reuse_a=*/false, /*reuse_b=*/false);
}

// A-fragment K index for element e (0..15) given lane_hi (lane>>4):
//   K = lane_hi*16 + e     [contiguous order, matches pack_b_kernel]
__device__ __forceinline__ int a_kidx(int e, int lanehi) {
  return (lanehi << 4) + e;
}

// ---------------------------------------------------------------------------
// generic utility kernels
// ---------------------------------------------------------------------------
__global__ void zero_f32_kernel(float* __restrict__ p, long n) {
  long i = (long)blockIdx.x * 256 + threadIdx.x;
  if (i < n) p[i] = 0.0f;
}

// copy x [NNODES,64] into padded buffer [NNODES,LD_X], zero pad
__global__ void pad_copy_kernel(const float* __restrict__ xin, float* __restrict__ xout) {
  long i = (long)blockIdx.x * 256 + threadIdx.x;
  long total = (long)NNODES * LD_X;
  if (i >= total) return;
  int col = (int)(i % LD_X);
  long row = i / LD_X;
  xout[i] = (col < 64) ? xin[row * 64 + col] : 0.0f;
}

// Pack row-major fp32 [K,N] (ld = row stride) into wave32 bf16 B-fragment layout:
//   dst[((kt*NT + nt)*32 + lane)*16 + e] = B[kt*32 + (lane>>4)*16 + e][nt*16 + (lane&15)]
__global__ void pack_b_kernel(const float* __restrict__ src, int ld, int K, int N,
                              int KT, int NT, __bf16* __restrict__ dst) {
  long total = (long)KT * NT * 512;
  long gid = (long)blockIdx.x * 256 + threadIdx.x;
  if (gid >= total) return;
  int e    = (int)(gid & 15);
  int lane = (int)((gid >> 4) & 31);
  long t   = gid >> 9;
  int nt   = (int)(t % NT);
  int kt   = (int)(t / NT);
  int k = kt * 32 + (lane >> 4) * 16 + e;
  int n = nt * 16 + (lane & 15);
  float v = (k < K && n < N) ? src[(long)k * ld + n] : 0.0f;
  dst[gid] = (__bf16)v;
}

// ---------------------------------------------------------------------------
// GraphConv: edge scatter  agg[dst] += x[src] * w
// ---------------------------------------------------------------------------
__global__ void edge_scatter_kernel(const int* __restrict__ src, const int* __restrict__ dst,
                                    const float* __restrict__ ew, const float* __restrict__ x,
                                    float* __restrict__ agg, int dimP) {
  int e = blockIdx.x * 256 + threadIdx.x;
  if (e >= NEDGES) return;
  int s = src[e], d = dst[e];
  float w = ew[e];
  const float* xs = x + (long)s * LD_X;
  float* ad = agg + (long)d * LD_X;
#pragma unroll 4
  for (int f = 0; f < dimP; ++f) atomicAdd(&ad[f], xs[f] * w);
}

// ---------------------------------------------------------------------------
// GraphConv node GEMM: xout = sigmoid(agg@Wr + b + x@Wo), WMMA bf16.
// Block = 256 threads = 8 waves; wave w owns 16 node rows. NTP n-tiles in regs.
// ---------------------------------------------------------------------------
template <int NTP>
__global__ void node_gemm_kernel(const float* __restrict__ agg, const float* __restrict__ x,
                                 const __bf16* __restrict__ pWr, const __bf16* __restrict__ pWo,
                                 const float* __restrict__ bias, float* __restrict__ xout,
                                 int out_dim, int KT) {
  int wave = threadIdx.x >> 5;
  int lane = threadIdx.x & 31;
  int lanelo = lane & 15, lanehi = lane >> 4;
  long row0 = (long)blockIdx.x * 128 + wave * 16;
  long rowA = row0 + lanelo;            // A-fragment source row for this lane
  bool rok = rowA < NNODES;

  v8f acc[NTP];
#pragma unroll
  for (int t = 0; t < NTP; ++t) acc[t] = (v8f){0,0,0,0,0,0,0,0};

  for (int kt = 0; kt < KT; ++kt) {
    long rbase = rowA * LD_X + kt * 32;
    v16bf aAgg, aX;
#pragma unroll
    for (int e = 0; e < 16; ++e) {
      int k = a_kidx(e, lanehi);
      aAgg[e] = (__bf16)(rok ? agg[rbase + k] : 0.0f);
      aX[e]   = (__bf16)(rok ? x[rbase + k]   : 0.0f);
    }
#pragma unroll
    for (int nt = 0; nt < NTP; ++nt) {
      const v16bf* br = (const v16bf*)(pWr + ((long)(kt * NTP + nt) * 32 + lane) * 16);
      const v16bf* bo = (const v16bf*)(pWo + ((long)(kt * NTP + nt) * 32 + lane) * 16);
      acc[nt] = wmma_bf16(aAgg, br[0], acc[nt]);
      acc[nt] = wmma_bf16(aX,   bo[0], acc[nt]);
    }
  }

#pragma unroll
  for (int nt = 0; nt < NTP; ++nt) {
    v8f a = acc[nt];
#pragma unroll
    for (int v = 0; v < 8; ++v) {
      long row = row0 + v + lanehi * 8;
      int col = nt * 16 + lanelo;
      if (row < NNODES) {
        float val = (col < out_dim) ? sigmoidf_(a[v] + bias[col]) : 0.0f;
        xout[row * LD_X + col] = val;
      }
    }
  }
}

// ---------------------------------------------------------------------------
// BatchNorm statistics: sums[c], sums[CORE+c] = sum, sumsq over nodes
// ---------------------------------------------------------------------------
__global__ void bn_stats_kernel(const float* __restrict__ x, float* __restrict__ sums) {
  int col = threadIdx.x;                 // blockDim = 160
  long r0 = (long)blockIdx.x * 250;
  float s = 0.0f, sq = 0.0f;
  for (long r = r0; r < r0 + 250 && r < NNODES; ++r) {
    float v = x[r * LD_X + col];
    s += v; sq += v * v;
  }
  atomicAdd(&sums[col], s);
  atomicAdd(&sums[CORE + col], sq);
}

// sel[w][c] = BN(x[waypoints[w]][c])
__global__ void sel_kernel(const float* __restrict__ x, const int* __restrict__ wp,
                           const float* __restrict__ sums, const float* __restrict__ gamma,
                           const float* __restrict__ beta, float* __restrict__ sel) {
  int gid = blockIdx.x * 256 + threadIdx.x;
  if (gid >= NWP * CORE) return;
  int c = gid % CORE, w = gid / CORE;
  float mean = sums[c] * (1.0f / NNODES);
  float var  = sums[CORE + c] * (1.0f / NNODES) - mean * mean;
  float v = x[(long)wp[w] * LD_X + c];
  sel[gid] = (v - mean) * rsqrtf(var + 1e-5f) * gamma[c] + beta[c];
}

// ---------------------------------------------------------------------------
// P/Q GEMM: out[256,2048] = sel[256,160] @ W (+ bias).  WMMA bf16, KT=5.
// Block = 8 waves; wave w computes n-tile (blockIdx.y*8 + w) of m-tile blockIdx.x
// ---------------------------------------------------------------------------
__global__ void pq_kernel(const float* __restrict__ sel, const __bf16* __restrict__ packB,
                          const float* __restrict__ bias, int has_bias,
                          float* __restrict__ out) {
  __shared__ float sA[16][CORE];
  int tid = threadIdx.x;
  for (int idx = tid; idx < 16 * CORE; idx += 256) {
    int r = idx / CORE, c = idx % CORE;
    sA[r][c] = sel[(blockIdx.x * 16 + r) * CORE + c];
  }
  __syncthreads();

  int wave = tid >> 5, lane = tid & 31, lanelo = lane & 15, lanehi = lane >> 4;
  int nt = blockIdx.y * 8 + wave;        // 0..127
  v8f acc = (v8f){0,0,0,0,0,0,0,0};
  for (int kt = 0; kt < 5; ++kt) {       // K = 160 = 5*32 exactly
    v16bf a;
#pragma unroll
    for (int e = 0; e < 16; ++e) {
      int k = kt * 32 + a_kidx(e, lanehi);
      a[e] = (__bf16)sA[lanelo][k];
    }
    const v16bf* bp = (const v16bf*)(packB + ((long)(kt * 128 + nt) * 32 + lane) * 16);
    acc = wmma_bf16(a, bp[0], acc);
  }
#pragma unroll
  for (int v = 0; v < 8; ++v) {
    int row = blockIdx.x * 16 + v + lanehi * 8;
    int col = nt * 16 + lanelo;
    float b = has_bias ? bias[col] : 0.0f;
    out[(long)row * HID + col] = acc[v] + b;
  }
}

// ---------------------------------------------------------------------------
// Pair kernel (hot GEMM): for pair (i,j): h = sigmoid(P[i]+Q[j]) [2048],
// o = h @ fc2_w (+fc2_b), res = sigmoid(max_n o), out[j*256+i] = res (0 on diag).
// Block = 8 waves, covers i-tile of 16 x j-block of 8; wave w owns j = j0+w.
// ---------------------------------------------------------------------------
__global__ void pair_kernel(const float* __restrict__ P, const float* __restrict__ Q,
                            const __bf16* __restrict__ packB2, const float* __restrict__ b2,
                            float* __restrict__ out) {
  __shared__ float sP[16][KCH];
  __shared__ float sQ[8][KCH];
  int tid = threadIdx.x, wave = tid >> 5, lane = tid & 31;
  int lanelo = lane & 15, lanehi = lane >> 4;
  int i0 = blockIdx.x * 16, j0 = blockIdx.y * 8;

  v8f acc[7];
#pragma unroll
  for (int t = 0; t < 7; ++t) acc[t] = (v8f){0,0,0,0,0,0,0,0};

  for (int ch = 0; ch < HID / KCH; ++ch) {
    __syncthreads();
    for (int idx = tid; idx < 16 * KCH; idx += 256) {
      int r = idx / KCH, c = idx % KCH;
      sP[r][c] = P[(long)(i0 + r) * HID + ch * KCH + c];
    }
    for (int idx = tid; idx < 8 * KCH; idx += 256) {
      int r = idx / KCH, c = idx % KCH;
      sQ[r][c] = Q[(long)(j0 + r) * HID + ch * KCH + c];
    }
    if (ch + 1 < HID / KCH) {  // emits global_prefetch_b8 for next P chunk
      __builtin_prefetch(&P[(long)(i0 + (tid & 15)) * HID + (ch + 1) * KCH + (tid >> 4) * 16], 0, 1);
    }
    __syncthreads();

    for (int ks = 0; ks < KCH / 32; ++ks) {
      v16bf a;
#pragma unroll
      for (int e = 0; e < 16; ++e) {
        int k = ks * 32 + a_kidx(e, lanehi);
        float h = sP[lanelo][k] + sQ[wave][k];
        a[e] = (__bf16)sigmoidf_(h);      // fused activation into bf16 convert
      }
      int kt = ch * (KCH / 32) + ks;
#pragma unroll
      for (int nt = 0; nt < 7; ++nt) {
        const v16bf* bp = (const v16bf*)(packB2 + ((long)(kt * 7 + nt) * 32 + lane) * 16);
        acc[nt] = wmma_bf16(a, bp[0], acc[nt]);
      }
    }
  }

  // max over N (cols 0..99), sigmoid, zero diagonal, store transposed
#pragma unroll
  for (int v = 0; v < 8; ++v) {
    float m = -3.4e38f;
#pragma unroll
    for (int nt = 0; nt < 7; ++nt) {
      int col = nt * 16 + lanelo;
      if (col < 100) m = fmaxf(m, acc[nt][v] + b2[col]);
    }
    for (int s = 1; s < 16; s <<= 1)           // reduce within 16-lane half (wave32)
      m = fmaxf(m, __shfl_xor(m, s, 32));
    if (lanelo == 0) {
      int i = i0 + v + lanehi * 8;
      int j = j0 + wave;
      float r = sigmoidf_(m);
      if (i == j) r = 0.0f;
      out[(long)j * NWP + i] = r;              // out[0][j][i] = pair(i, j)
    }
  }
}

// ---------------------------------------------------------------------------
// host-side launch
// ---------------------------------------------------------------------------
static void launch_node(int NTP, const float* agg, const float* x, const __bf16* pWr,
                        const __bf16* pWo, const float* bias, float* xout,
                        int out_dim, int KT, hipStream_t s) {
  dim3 g((NNODES + 127) / 128), b(256);
  switch (NTP) {
    case 2:  node_gemm_kernel<2> <<<g, b, 0, s>>>(agg, x, pWr, pWo, bias, xout, out_dim, KT); break;
    case 4:  node_gemm_kernel<4> <<<g, b, 0, s>>>(agg, x, pWr, pWo, bias, xout, out_dim, KT); break;
    case 6:  node_gemm_kernel<6> <<<g, b, 0, s>>>(agg, x, pWr, pWo, bias, xout, out_dim, KT); break;
    case 8:  node_gemm_kernel<8> <<<g, b, 0, s>>>(agg, x, pWr, pWo, bias, xout, out_dim, KT); break;
    default: node_gemm_kernel<10><<<g, b, 0, s>>>(agg, x, pWr, pWo, bias, xout, out_dim, KT); break;
  }
}

extern "C" void kernel_launch(void* const* d_in, const int* in_sizes, int n_in,
                              void* d_out, int out_size, void* d_ws, size_t ws_size,
                              hipStream_t stream) {
  (void)in_sizes; (void)n_in; (void)out_size; (void)ws_size;
  // input order: x, edge_index, edge_w, waypoints, w_rel[0..10], w_root[0..10],
  //              b_rel[0..10], bn_gamma, bn_beta, fc1_w, fc1_b, fc2_w, fc2_b
  const float* x_in   = (const float*)d_in[0];
  const int*   eidx   = (const int*)d_in[1];
  const float* ew     = (const float*)d_in[2];
  const int*   wp     = (const int*)d_in[3];
  const float* gamma  = (const float*)d_in[37];
  const float* beta   = (const float*)d_in[38];
  const float* fc1_w  = (const float*)d_in[39];
  const float* fc1_b  = (const float*)d_in[40];
  const float* fc2_w  = (const float*)d_in[41];
  const float* fc2_b  = (const float*)d_in[42];
  const int* esrc = eidx;
  const int* edst = eidx + NEDGES;

  // deterministic bump allocator over workspace
  size_t off = 0;
  auto alloc = [&](size_t bytes) -> char* {
    char* p = (char*)d_ws + off;
    off += (bytes + 255) & ~(size_t)255;
    return p;
  };
  const size_t XB = (size_t)NNODES * LD_X * 4;
  float*  xA   = (float*)alloc(XB);
  float*  xB   = (float*)alloc(XB);
  float*  agg  = (float*)alloc(XB);
  float*  sums = (float*)alloc(2 * CORE * 4);
  float*  sel  = (float*)alloc((size_t)NWP * CORE * 4);
  float*  Pm   = (float*)alloc((size_t)NWP * HID * 4);
  float*  Qm   = (float*)alloc((size_t)NWP * HID * 4);
  __bf16* pF1t = (__bf16*)alloc((size_t)5 * 128 * 512 * 2);
  __bf16* pF1b = (__bf16*)alloc((size_t)5 * 128 * 512 * 2);
  __bf16* pF2  = (__bf16*)alloc((size_t)64 * 7 * 512 * 2);

  const int dims[12] = {64, 10, 25, 40, 55, 70, 85, 100, 115, 130, 145, 160};
  __bf16* pWr[11]; __bf16* pWo[11];
  for (int l = 0; l < 11; ++l) {
    int in = dims[l], out = dims[l + 1];
    int KT = (in + 31) / 32, NTP = ((out + 31) / 32) * 2;
    size_t bytes = (size_t)KT * NTP * 512 * 2;
    pWr[l] = (__bf16*)alloc(bytes);
    pWo[l] = (__bf16*)alloc(bytes);
  }

  // 1) pad input features into xA
  {
    long n = (long)NNODES * LD_X;
    pad_copy_kernel<<<dim3((unsigned)((n + 255) / 256)), 256, 0, stream>>>(x_in, xA);
  }

  // 2) 11 GraphConv layers
  float* xcur = xA; float* xnxt = xB;
  for (int l = 0; l < 11; ++l) {
    int in = dims[l], out = dims[l + 1];
    int KT = (in + 31) / 32, NTP = ((out + 31) / 32) * 2;
    int dimP = (in + 3) & ~3;
    // pack weights (bf16 B-fragment layout)
    long tpack = (long)KT * NTP * 512;
    pack_b_kernel<<<dim3((unsigned)((tpack + 255) / 256)), 256, 0, stream>>>(
        (const float*)d_in[4 + l], out, in, out, KT, NTP, pWr[l]);
    pack_b_kernel<<<dim3((unsigned)((tpack + 255) / 256)), 256, 0, stream>>>(
        (const float*)d_in[15 + l], out, in, out, KT, NTP, pWo[l]);
    // zero aggregation buffer, scatter edges, node GEMM
    long n = (long)NNODES * LD_X;
    zero_f32_kernel<<<dim3((unsigned)((n + 255) / 256)), 256, 0, stream>>>(agg, n);
    edge_scatter_kernel<<<dim3((NEDGES + 255) / 256), 256, 0, stream>>>(
        esrc, edst, ew, xcur, agg, dimP);
    launch_node(NTP, agg, xcur, pWr[l], pWo[l], (const float*)d_in[26 + l],
                xnxt, out, KT, stream);
    float* t = xcur; xcur = xnxt; xnxt = t;
  }

  // 3) BatchNorm stats + waypoint selection
  zero_f32_kernel<<<1, 256, 0, stream>>>(sums, 2 * CORE);
  bn_stats_kernel<<<dim3(200), 160, 0, stream>>>(xcur, sums);
  sel_kernel<<<dim3((NWP * CORE + 255) / 256), 256, 0, stream>>>(xcur, wp, sums, gamma, beta, sel);

  // 4) P = sel @ fc1_w[:160] + fc1_b ; Q = sel @ fc1_w[160:]
  pack_b_kernel<<<dim3((5 * 128 * 512 + 255) / 256), 256, 0, stream>>>(
      fc1_w, HID, CORE, HID, 5, 128, pF1t);
  pack_b_kernel<<<dim3((5 * 128 * 512 + 255) / 256), 256, 0, stream>>>(
      fc1_w + (size_t)CORE * HID, HID, CORE, HID, 5, 128, pF1b);
  pq_kernel<<<dim3(16, 16), 256, 0, stream>>>(sel, pF1t, fc1_b, 1, Pm);
  pq_kernel<<<dim3(16, 16), 256, 0, stream>>>(sel, pF1b, fc1_b, 0, Qm);

  // 5) pack fc2_w, run the pair GEMM + max + sigmoid + transpose epilogue
  pack_b_kernel<<<dim3((64 * 7 * 512 + 255) / 256), 256, 0, stream>>>(
      fc2_w, 100, HID, 100, 64, 7, pF2);
  pair_kernel<<<dim3(16, 32), 256, 0, stream>>>(Pm, Qm, pF2, fc2_b, (float*)d_out);
}
